// Qwen3MoeSparseMoeBlockPatch_84078279786501
// MI455X (gfx1250) — matrix operand order repaired and checked by reading, not compiled, not run on hardware
//
#include <hip/hip_runtime.h>
#include <math.h>

// ---------------- problem constants ----------------
#define T_TOK   4096      // B*S tokens
#define HID     2048
#define INTR    768
#define NEXP    64
#define TOPK    8
#define NPAIR   (T_TOK * TOPK)        // 32768
#define MT      128                   // M tile (pairs) per workgroup
#define PERM_CAP (NPAIR + NEXP * MT)  // 40960 padded perm slots
#define MAX_TILES (PERM_CAP / MT)     // 320

// ---------------- workspace layout (bytes) ----------------
static const size_t OFF_TOPK_IDX = 0;
static const size_t OFF_TOPK_W   = OFF_TOPK_IDX + (size_t)NPAIR * 4;
static const size_t OFF_COUNTS   = OFF_TOPK_W   + (size_t)NPAIR * 4;
static const size_t OFF_CURSOR   = OFF_COUNTS   + 256;
static const size_t OFF_OFFSETS  = OFF_CURSOR   + 256;
static const size_t OFF_TILE_E   = OFF_OFFSETS  + 512;
static const size_t OFF_TILE_M   = OFF_TILE_E   + 2048;
static const size_t OFF_TOTAL    = OFF_TILE_M   + 2048;
static const size_t OFF_PERM     = OFF_TOTAL    + 256;
static const size_t OFF_ACT      = ((OFF_PERM + (size_t)PERM_CAP * 4 + 255) / 256) * 256;

// ---------------- types ----------------
typedef __bf16 bf16;
typedef __attribute__((ext_vector_type(16))) __bf16 v16bf;
typedef __attribute__((ext_vector_type(8)))  __bf16 v8bf;
typedef __attribute__((ext_vector_type(4)))  __bf16 v4bf;
typedef __attribute__((ext_vector_type(8)))  float  v8f;
typedef __attribute__((ext_vector_type(4)))  float  v4f;

// A fragment for v_wmma_f32_16x16x32_bf16 from LDS tile [rows][32] bf16:
// lane<16 : row=lane,    K = 0..15
// lane>=16: row=lane-16, K = 16..31
__device__ __forceinline__ v16bf load_a_frag(const bf16* sA, int row, int klo) {
  return *(const v16bf*)(sA + row * 32 + klo);
}

// B fragment: lane<16: col=lane, K=0..15 ; lane>=16: col=lane-16, K=16..31.
// Weights staged TRANSPOSED in LDS as Wt[n][k] so 16 K values are contiguous.
__device__ __forceinline__ v16bf load_b_frag(const bf16* sW, int col, int klo) {
  return *(const v16bf*)(sW + col * 32 + klo);
}

__device__ __forceinline__ float silu(float g) {
  return g / (1.0f + __expf(-g));
}

// ---------------- kernel 0: zero the combine output region ----------------
__global__ void kzero_out(float* __restrict__ out) {
  const size_t n = (size_t)T_TOK * HID;
  for (size_t i = (size_t)blockIdx.x * blockDim.x + threadIdx.x; i < n;
       i += (size_t)gridDim.x * blockDim.x)
    out[i] = 0.0f;
}

// ---------------- kernel 1: init routing state ----------------
__global__ void kinit_route(int* __restrict__ perm, int* __restrict__ counts,
                            int* __restrict__ cursor) {
  int i = blockIdx.x * blockDim.x + threadIdx.x;
  if (i < PERM_CAP) perm[i] = -1;
  if (i < NEXP) { counts[i] = 0; cursor[i] = 0; }
}

// ---------------- kernel 2: router logits (exact fp32) ----------------
__global__ void krouter(const float* __restrict__ x,
                        const float* __restrict__ gate_w,
                        float* __restrict__ logits) {
  __shared__ float xs[HID];
  const int t = blockIdx.x;
  const float* xr = x + (size_t)t * HID;
#pragma unroll
  for (int i = 0; i < HID / 64; ++i) xs[threadIdx.x + i * 64] = xr[threadIdx.x + i * 64];
  __syncthreads();
  const int e = threadIdx.x;
  const float* wr = gate_w + (size_t)e * HID;
  float acc = 0.f;
#pragma unroll 4
  for (int h = 0; h < HID; ++h) acc = fmaf(xs[h], wr[h], acc);
  logits[(size_t)t * NEXP + e] = acc;
}

// ---------------- kernel 3: softmax + top-8 + renorm + counts --------------
__global__ void ktopk(const float* __restrict__ logits,
                      int* __restrict__ topk_idx, float* __restrict__ topk_w,
                      int* __restrict__ counts) {
  const int t = blockIdx.x * blockDim.x + threadIdx.x;
  if (t >= T_TOK) return;
  const float* lg = logits + (size_t)t * NEXP;
  float mx = lg[0];
  for (int e = 1; e < NEXP; ++e) mx = fmaxf(mx, lg[e]);
  float s = 0.f;
  for (int e = 0; e < NEXP; ++e) s += __expf(lg[e] - mx);
  unsigned long long used = 0ull;
  int   sel[TOPK];
  float wv[TOPK];
  float wsum = 0.f;
#pragma unroll
  for (int k = 0; k < TOPK; ++k) {
    int best = 0; float bv = -3.4e38f;
    for (int e = 0; e < NEXP; ++e) {
      if ((used >> e) & 1ull) continue;
      float v = lg[e];
      if (v > bv) { bv = v; best = e; }   // strict > : lowest index wins ties
    }
    used |= 1ull << best;
    float p = __expf(bv - mx) / s;
    sel[k] = best; wv[k] = p; wsum += p;
  }
  const float inv = 1.0f / wsum;
#pragma unroll
  for (int k = 0; k < TOPK; ++k) {
    topk_idx[t * TOPK + k] = sel[k];
    topk_w[t * TOPK + k]   = wv[k] * inv;
    atomicAdd(&counts[sel[k]], 1);
  }
}

// ---------------- kernel 4: offsets scan + tile table ----------------------
__global__ void kscan(const int* __restrict__ counts, int* __restrict__ offsets,
                      int* __restrict__ tile_e, int* __restrict__ tile_m,
                      int* __restrict__ total) {
  if (threadIdx.x != 0 || blockIdx.x != 0) return;
  int off = 0, tc = 0;
  for (int e = 0; e < NEXP; ++e) {
    offsets[e] = off;
    int padded = ((counts[e] + MT - 1) / MT) * MT;
    for (int j = 0; j < padded / MT; ++j) {
      tile_e[tc] = e;
      tile_m[tc] = off + j * MT;
      ++tc;
    }
    off += padded;
  }
  offsets[NEXP] = off;
  total[0] = tc;
}

// ---------------- kernel 5: build grouped (padded) pair list ---------------
__global__ void kperm(const int* __restrict__ topk_idx,
                      const int* __restrict__ offsets,
                      int* __restrict__ cursor, int* __restrict__ perm) {
  const int p = blockIdx.x * blockDim.x + threadIdx.x;
  if (p >= NPAIR) return;
  const int e = topk_idx[p];
  const int slot = atomicAdd(&cursor[e], 1);
  perm[offsets[e] + slot] = p;
}

// ---------------- kernel 6: up+gate GEMM + SiLU (bf16 WMMA, fp32 acc) ------
// grid=(MAX_TILES, INTR/64), block=256 (8 waves x 16 rows x 64 cols), LDS 2x buffered
__global__ void __launch_bounds__(256)
kgemm_upgate(const float* __restrict__ x, const float* __restrict__ w1,
             const float* __restrict__ w2, const int* __restrict__ tile_e,
             const int* __restrict__ tile_m, const int* __restrict__ total,
             const int* __restrict__ perm, bf16* __restrict__ act) {
  if ((int)blockIdx.x >= total[0]) return;
  const int e  = tile_e[blockIdx.x];
  const int mb = tile_m[blockIdx.x];
  const int nb = blockIdx.y * 64;
  const int NKB = HID / 32;

  __shared__ __align__(32) bf16 sA[2][MT * 32];   // 2 x 8 KB
  __shared__ __align__(32) bf16 sW1[2][64 * 32];  // 2 x 4 KB (transposed [n][k])
  __shared__ __align__(32) bf16 sW2[2][64 * 32];  // 2 x 4 KB
  __shared__ int pairsS[MT];

  if (threadIdx.x < MT) pairsS[threadIdx.x] = perm[mb + threadIdx.x];

  const int wave  = threadIdx.x >> 5;
  const int lane  = threadIdx.x & 31;
  const int lhalf = lane >> 4;
  const int l15   = lane & 15;

  const size_t wbase = (size_t)e * HID * INTR;

  // ---- per-thread gather state hoisted into registers ----
  const int kq = ((int)threadIdx.x & 7) * 4;   // fixed K-quad within the 32-wide tile
  const int r0 = (int)threadIdx.x >> 3;        // rows r0, r0+32, r0+64, r0+96
  const float* xp[4];
  bool inval[4];
#pragma unroll
  for (int i = 0; i < 4; ++i) {
    int p = perm[mb + r0 + i * 32];
    inval[i] = (p < 0);
    int tok = inval[i] ? 0 : (p >> 3);
    xp[i] = x + (size_t)tok * HID + kq;
  }
  // weight-staging coords (pure arithmetic)
  const int wk0 = (int)threadIdx.x >> 4;          // k of quad 0 (0..15); quad 1 at +16
  const int wnq = ((int)threadIdx.x & 15) * 4;    // n quad

  // ---- fused staging: issue ALL global loads, then convert+store ----
  auto stage = [&](int kb, int buf) {
    // load phase: 4 A-row quads + 2x2 weight quads, all independent
    v4f va[4], wa[2], wc[2];
#pragma unroll
    for (int i = 0; i < 4; ++i) va[i] = *(const v4f*)(xp[i] + kb * 32);
#pragma unroll
    for (int i = 0; i < 2; ++i) {
      int k = wk0 + i * 16;
      size_t g = wbase + (size_t)(kb * 32 + k) * INTR + nb + wnq;
      wa[i] = *(const v4f*)(w1 + g);
      wc[i] = *(const v4f*)(w2 + g);
    }
    // convert + LDS store phase
#pragma unroll
    for (int i = 0; i < 4; ++i) {
      v4bf b;
#pragma unroll
      for (int j = 0; j < 4; ++j) b[j] = (bf16)(inval[i] ? 0.f : va[i][j]);
      *(v4bf*)&sA[buf][(r0 + i * 32) * 32 + kq] = b;
    }
#pragma unroll
    for (int i = 0; i < 2; ++i) {
      int k = wk0 + i * 16;
#pragma unroll
      for (int j = 0; j < 4; ++j) {
        sW1[buf][(wnq + j) * 32 + k] = (bf16)wa[i][j];
        sW2[buf][(wnq + j) * 32 + k] = (bf16)wc[i][j];
      }
    }
    // prefetch two K-steps ahead (global_prefetch_b8)
    if (kb + 2 < NKB) {
      __builtin_prefetch(xp[0] + (kb + 2) * 32, 0, 1);
      size_t g2 = wbase + (size_t)((kb + 2) * 32 + wk0) * INTR + nb + wnq;
      __builtin_prefetch(w1 + g2, 0, 1);
      __builtin_prefetch(w2 + g2, 0, 1);
    }
  };

  const v8f z = {0.f, 0.f, 0.f, 0.f, 0.f, 0.f, 0.f, 0.f};
  v8f accU[4] = {z, z, z, z};
  v8f accG[4] = {z, z, z, z};

  stage(0, 0);
  for (int kb = 0; kb < NKB; ++kb) {
    const int cur = kb & 1;
    __syncthreads();
    if (kb + 1 < NKB) stage(kb + 1, cur ^ 1);

    v16bf af = load_a_frag(&sA[cur][0], wave * 16 + l15, lhalf * 16);
    v16bf b1[4], b2[4];
#pragma unroll
    for (int t = 0; t < 4; ++t) {
      b1[t] = load_b_frag(&sW1[cur][0], t * 16 + l15, lhalf * 16);
      b2[t] = load_b_frag(&sW2[cur][0], t * 16 + l15, lhalf * 16);
    }
#pragma unroll
    for (int t = 0; t < 4; ++t)
      accU[t] = __builtin_amdgcn_wmma_f32_16x16x32_bf16(
          false, af, false, b1[t], (short)0, accU[t], false, false);
#pragma unroll
    for (int t = 0; t < 4; ++t)
      accG[t] = __builtin_amdgcn_wmma_f32_16x16x32_bf16(
          false, af, false, b2[t], (short)0, accG[t], false, false);
  }

  // act = silu(gate) * up -> bf16 scratch indexed by pair id
#pragma unroll
  for (int t = 0; t < 4; ++t) {
#pragma unroll
    for (int j = 0; j < 8; ++j) {
      int m = wave * 16 + j + lhalf * 8;   // C layout: VGPR j -> M=j (+8 hi half)
      int p = pairsS[m];
      if (p >= 0) {
        float a = accU[t][j] * silu(accG[t][j]);
        act[(size_t)p * INTR + nb + t * 16 + l15] = (bf16)a;
      }
    }
  }
}

// ---------------- kernel 7: down GEMM + weighted atomic combine ------------
// grid=(MAX_TILES, HID/128), block=256 (8 waves x 16 rows x 128 cols), 2x buffered
__global__ void __launch_bounds__(256)
kgemm_down(const bf16* __restrict__ act, const float* __restrict__ w3,
           const float* __restrict__ topk_w, const int* __restrict__ tile_e,
           const int* __restrict__ tile_m, const int* __restrict__ total,
           const int* __restrict__ perm, float* __restrict__ out) {
  if ((int)blockIdx.x >= total[0]) return;
  const int e  = tile_e[blockIdx.x];
  const int mb = tile_m[blockIdx.x];
  const int nb = blockIdx.y * 128;
  const int NKB = INTR / 32;

  __shared__ __align__(32) bf16 sA[2][MT * 32];    // 2 x 8 KB
  __shared__ __align__(32) bf16 sW3[2][128 * 32];  // 2 x 8 KB (transposed [n][k])
  __shared__ int   pairsS[MT];
  __shared__ float wS[MT];

  if (threadIdx.x < MT) {
    int p = perm[mb + threadIdx.x];
    pairsS[threadIdx.x] = p;
    wS[threadIdx.x] = (p >= 0) ? topk_w[p] : 0.f;
  }

  const int wave  = threadIdx.x >> 5;
  const int lane  = threadIdx.x & 31;
  const int lhalf = lane >> 4;
  const int l15   = lane & 15;

  const size_t wbase = (size_t)e * INTR * HID;

  // ---- per-thread gather state in registers ----
  const int ko  = ((int)threadIdx.x & 3) * 8;  // fixed 8-elem bf16 chunk
  const int r0d = (int)threadIdx.x >> 2;       // rows r0d, r0d+64
  const bf16* ap[2];
  bool invalD[2];
#pragma unroll
  for (int i = 0; i < 2; ++i) {
    int p = perm[mb + r0d + i * 64];
    invalD[i] = (p < 0);
    int pc = invalD[i] ? 0 : p;
    ap[i] = act + (size_t)pc * INTR + ko;
  }
  const int wk0 = (int)threadIdx.x >> 5;        // k of quad 0 (0..7); quads at +8,+16,+24
  const int wnq = ((int)threadIdx.x & 31) * 4;  // n quad

  auto stage = [&](int kb, int buf) {
    // load phase: 2 A-octets + 4 weight quads
    v8bf av[2];
    v4f  wv4[4];
#pragma unroll
    for (int i = 0; i < 2; ++i) av[i] = *(const v8bf*)(ap[i] + kb * 32);
#pragma unroll
    for (int i = 0; i < 4; ++i) {
      int k = wk0 + i * 8;
      size_t g = wbase + (size_t)(kb * 32 + k) * HID + nb + wnq;
      wv4[i] = *(const v4f*)(w3 + g);
    }
    // store phase
#pragma unroll
    for (int i = 0; i < 2; ++i) {
      v8bf v = av[i];
#pragma unroll
      for (int j = 0; j < 8; ++j) if (invalD[i]) v[j] = (bf16)0.f;
      *(v8bf*)&sA[buf][(r0d + i * 64) * 32 + ko] = v;
    }
#pragma unroll
    for (int i = 0; i < 4; ++i) {
      int k = wk0 + i * 8;
#pragma unroll
      for (int j = 0; j < 4; ++j) sW3[buf][(wnq + j) * 32 + k] = (bf16)wv4[i][j];
    }
    if (kb + 2 < NKB) {
      __builtin_prefetch(ap[0] + (kb + 2) * 32, 0, 1);
      size_t g2 = wbase + (size_t)((kb + 2) * 32 + wk0) * HID + nb + wnq;
      __builtin_prefetch(w3 + g2, 0, 1);
    }
  };

  const v8f z = {0.f, 0.f, 0.f, 0.f, 0.f, 0.f, 0.f, 0.f};
  v8f accD[8] = {z, z, z, z, z, z, z, z};

  stage(0, 0);
  for (int kb = 0; kb < NKB; ++kb) {
    const int cur = kb & 1;
    __syncthreads();
    if (kb + 1 < NKB) stage(kb + 1, cur ^ 1);

    v16bf af = load_a_frag(&sA[cur][0], wave * 16 + l15, lhalf * 16);
    v16bf bb[8];
#pragma unroll
    for (int t = 0; t < 8; ++t)
      bb[t] = load_b_frag(&sW3[cur][0], t * 16 + l15, lhalf * 16);
#pragma unroll
    for (int t = 0; t < 8; ++t)
      accD[t] = __builtin_amdgcn_wmma_f32_16x16x32_bf16(
          false, af, false, bb[t], (short)0, accD[t], false, false);
  }

  // weighted combine into out[token]
#pragma unroll
  for (int t = 0; t < 8; ++t) {
#pragma unroll
    for (int j = 0; j < 8; ++j) {
      int m = wave * 16 + j + lhalf * 8;
      int p = pairsS[m];
      if (p >= 0) {
        int tok = p >> 3;
        float v = accD[t][j] * wS[m];
        atomicAdd(&out[(size_t)tok * HID + nb + t * 16 + l15], v);
      }
    }
  }
}

// ---------------- launch ----------------
extern "C" void kernel_launch(void* const* d_in, const int* in_sizes, int n_in,
                              void* d_out, int out_size, void* d_ws, size_t ws_size,
                              hipStream_t stream) {
  (void)in_sizes; (void)n_in; (void)out_size; (void)ws_size;
  const float* x      = (const float*)d_in[0];
  const float* gate_w = (const float*)d_in[1];
  const float* w1     = (const float*)d_in[2];  // [E,H,I] up
  const float* w2     = (const float*)d_in[3];  // [E,H,I] gate
  const float* w3     = (const float*)d_in[4];  // [E,I,H] down

  float* out    = (float*)d_out;                 // [T,H]
  float* logits = out + (size_t)T_TOK * HID;     // [T,E] (2nd output)

  char* wsb = (char*)d_ws;
  int*   topk_idx = (int*)(wsb + OFF_TOPK_IDX);
  float* topk_w   = (float*)(wsb + OFF_TOPK_W);
  int*   counts   = (int*)(wsb + OFF_COUNTS);
  int*   cursor   = (int*)(wsb + OFF_CURSOR);
  int*   offsets  = (int*)(wsb + OFF_OFFSETS);
  int*   tile_e   = (int*)(wsb + OFF_TILE_E);
  int*   tile_m   = (int*)(wsb + OFF_TILE_M);
  int*   total    = (int*)(wsb + OFF_TOTAL);
  int*   perm     = (int*)(wsb + OFF_PERM);
  bf16*  act      = (bf16*)(wsb + OFF_ACT);

  kzero_out<<<2048, 256, 0, stream>>>(out);
  kinit_route<<<(PERM_CAP + 255) / 256, 256, 0, stream>>>(perm, counts, cursor);
  krouter<<<T_TOK, 64, 0, stream>>>(x, gate_w, logits);
  ktopk<<<(T_TOK + 255) / 256, 256, 0, stream>>>(logits, topk_idx, topk_w, counts);
  kscan<<<1, 1, 0, stream>>>(counts, offsets, tile_e, tile_m, total);
  kperm<<<(NPAIR + 255) / 256, 256, 0, stream>>>(topk_idx, offsets, cursor, perm);
  kgemm_upgate<<<dim3(MAX_TILES, INTR / 64), 256, 0, stream>>>(
      x, w1, w2, tile_e, tile_m, total, perm, act);
  kgemm_down<<<dim3(MAX_TILES, HID / 128), 256, 0, stream>>>(
      act, w3, topk_w, tile_e, tile_m, total, perm, out);
}
